// Decoder_4681514353181
// MI455X (gfx1250) — matrix operand order repaired and checked by reading, not compiled, not run on hardware
//
#include <hip/hip_runtime.h>
#include <hip/hip_bf16.h>

typedef __attribute__((ext_vector_type(16))) __bf16 v16bf;
typedef __attribute__((ext_vector_type(8)))  __bf16 v8bf;
typedef __attribute__((ext_vector_type(8)))  float  v8f;

// Problem dims
#define TT 500
#define BB 64
#define KSD 128
#define VSD 128
#define HH 512
#define EE 256
#define VOC 4096
#define LL 250

// GEMM1: M=64, N=2048, K=896 (384 [emb|ctx] + 512 h1)
#define K1 896
#define N1 2048
#define NT1 128
#define KC1 28
// GEMM2: M=64, N=512, K=640 (512 h1 + 128 h2)
#define K2 640
#define N2 512
#define NT2 32
#define KC2 20
// GEMM out: M=64, N=4096, K=256 (128 h2 + 128 ctx)
#define K3 256
#define N3 4096
#define NT3 256
#define KC3 8

#define NWG 64
#define NTHR 256

// ---------------- workspace layout (bytes) ----------------
static constexpr size_t SZ_WF1 = (size_t)NT1 * KC1 * 32 * 16 * 2;  // bf16 frags gemm1
static constexpr size_t SZ_WF2 = (size_t)NT2 * KC2 * 32 * 16 * 2;
static constexpr size_t SZ_WF3 = (size_t)NT3 * KC3 * 32 * 16 * 2;
static constexpr size_t OFF_WF1 = 0;
static constexpr size_t OFF_WF2 = OFF_WF1 + SZ_WF1;
static constexpr size_t OFF_WF3 = OFF_WF2 + SZ_WF2;
static constexpr size_t OFF_B1  = OFF_WF3 + SZ_WF3;            // f32[2048]
static constexpr size_t OFF_B2  = OFF_B1 + 2048 * 4;           // f32[512]
static constexpr size_t OFF_Z1  = OFF_B2 + 512 * 4;            // bf16[64*896]
static constexpr size_t OFF_Z2  = OFF_Z1 + (size_t)BB * K1 * 2;// bf16[64*640]
static constexpr size_t OFF_ZO  = OFF_Z2 + (size_t)BB * K2 * 2;// bf16[64*256]
static constexpr size_t OFF_G1  = OFF_ZO + (size_t)BB * K3 * 2;// f32[64*2048]
static constexpr size_t OFF_G2  = OFF_G1 + (size_t)BB * N1 * 4;// f32[64*512]
static constexpr size_t OFF_C1  = OFF_G2 + (size_t)BB * N2 * 4;// f32[64*512]
static constexpr size_t OFF_C2  = OFF_C1 + (size_t)BB * HH * 4;// f32[64*128]
static constexpr size_t OFF_H2F = OFF_C2 + (size_t)BB * KSD * 4;// f32[64*128]
static constexpr size_t OFF_BAR = OFF_H2F + (size_t)BB * KSD * 4;// u32[2]

// ---------------- helpers ----------------
__device__ __forceinline__ __bf16 f2bf(float f) {
  unsigned u = __float_as_uint(f);
  unsigned r = (u + 0x7FFFu + ((u >> 16) & 1u)) >> 16;
  unsigned short s = (unsigned short)r;
  return *(__bf16*)&s;
}

__device__ __forceinline__ float sigf(float x) { return 1.0f / (1.0f + __expf(-x)); }

// A fragment (16x32 bf16): lane l -> row m0+(l&15),
// K = k0 + (l>=16 ? 16 : 0) + {0..15} (contiguous, matches B packing).
__device__ __forceinline__ v16bf load_a_frag(const __bf16* Z, int ldk, int m0, int k0, int lane) {
  int m  = m0 + (lane & 15);
  int kb = k0 + ((lane >> 4) << 4);
  const __bf16* p = Z + (size_t)m * ldk + kb;
  return *(const v16bf*)(p);
}

// one 16x16 output tile, K accumulated over kcN chunks of 32 (B pre-swizzled)
__device__ __forceinline__ void gemm_tile(const __bf16* Z, int ldk, const __bf16* wf,
                                          int nt, int kcN, int m0, int lane, v8f& acc) {
  const __bf16* bsrc = wf + (size_t)nt * kcN * 32 * 16 + (size_t)lane * 16;
  for (int kc = 0; kc < kcN; ++kc) {
    v16bf a  = load_a_frag(Z, ldk, m0, kc * 32, lane);
    v16bf bb = *(const v16bf*)(bsrc + (size_t)kc * 32 * 16);
    acc = __builtin_amdgcn_wmma_f32_16x16x32_bf16(false, a, false, bb, (short)0, acc, false, false);
  }
}

__device__ __forceinline__ void grid_barrier(unsigned* bar) {
  __syncthreads();
  if (threadIdx.x == 0) {
    __threadfence();
    volatile unsigned* gen = bar + 1;
    unsigned g = *gen;
    if (atomicAdd(bar, 1u) == NWG - 1u) {
      *(volatile unsigned*)bar = 0;
      __threadfence();
      atomicAdd((unsigned*)(bar + 1), 1u);
    } else {
      while (*gen == g) { __builtin_amdgcn_s_sleep(1); }
    }
    __threadfence();
  }
  __syncthreads();
}

// ---------------- prep: swizzle weights into WMMA-B fragment layout ----------------
__global__ void prep_kernel(const float* w_ih1, const float* w_hh1,
                            const float* w_ih2, const float* w_hh2,
                            const float* w_out,
                            const float* b_ih1, const float* b_hh1,
                            const float* b_ih2, const float* b_hh2,
                            char* ws) {
  __bf16* wf1 = (__bf16*)(ws + OFF_WF1);
  __bf16* wf2 = (__bf16*)(ws + OFF_WF2);
  __bf16* wf3 = (__bf16*)(ws + OFF_WF3);
  float* bias1 = (float*)(ws + OFF_B1);
  float* bias2 = (float*)(ws + OFF_B2);

  const size_t S1 = (size_t)NT1 * KC1 * 32;
  const size_t S2 = (size_t)NT2 * KC2 * 32;
  const size_t S3 = (size_t)NT3 * KC3 * 32;
  size_t gid = (size_t)blockIdx.x * blockDim.x + threadIdx.x;
  size_t stride = (size_t)gridDim.x * blockDim.x;

  for (size_t slot = gid; slot < S1 + S2 + S3; slot += stride) {
    if (slot < S1) {
      size_t t = slot / (KC1 * 32), rem = slot % (KC1 * 32);
      int chunk = (int)(rem / 32), lane = (int)(rem % 32);
      int n = (int)t * 16 + (lane & 15);
      int kb = chunk * 32 + ((lane >> 4) << 4);
      for (int e = 0; e < 16; ++e) {
        int k = kb + e;
        float w = (k < 384) ? w_ih1[(size_t)n * 384 + k] : w_hh1[(size_t)n * 512 + (k - 384)];
        wf1[slot * 16 + e] = f2bf(w);
      }
    } else if (slot < S1 + S2) {
      size_t s2 = slot - S1;
      size_t t = s2 / (KC2 * 32), rem = s2 % (KC2 * 32);
      int chunk = (int)(rem / 32), lane = (int)(rem % 32);
      int n = (int)t * 16 + (lane & 15);
      int kb = chunk * 32 + ((lane >> 4) << 4);
      for (int e = 0; e < 16; ++e) {
        int k = kb + e;
        float w = (k < 512) ? w_ih2[(size_t)n * 512 + k] : w_hh2[(size_t)n * 128 + (k - 512)];
        wf2[s2 * 16 + e] = f2bf(w);
      }
    } else {
      size_t s3 = slot - S1 - S2;
      size_t t = s3 / (KC3 * 32), rem = s3 % (KC3 * 32);
      int chunk = (int)(rem / 32), lane = (int)(rem % 32);
      int n = (int)t * 16 + (lane & 15);
      int kb = chunk * 32 + ((lane >> 4) << 4);
      for (int e = 0; e < 16; ++e) {
        int k = kb + e;
        wf3[s3 * 16 + e] = f2bf(w_out[(size_t)n * 256 + k]);
      }
    }
  }
  for (size_t i = gid; i < 2048; i += stride) bias1[i] = b_ih1[i] + b_hh1[i];
  for (size_t i = gid; i < 512; i += stride) bias2[i] = b_ih2[i] + b_hh2[i];
}

// ---------------- init: zero state, context = values[0] ----------------
__global__ void init_kernel(const float* values, char* ws) {
  float* c1f = (float*)(ws + OFF_C1);
  float* c2f = (float*)(ws + OFF_C2);
  float* h2f = (float*)(ws + OFF_H2F);
  __bf16* z1 = (__bf16*)(ws + OFF_Z1);
  __bf16* z2 = (__bf16*)(ws + OFF_Z2);
  unsigned* bar = (unsigned*)(ws + OFF_BAR);
  int gid = blockIdx.x * blockDim.x + threadIdx.x;
  int stride = gridDim.x * blockDim.x;
  for (int i = gid; i < BB * HH; i += stride) c1f[i] = 0.f;
  for (int i = gid; i < BB * KSD; i += stride) { c2f[i] = 0.f; h2f[i] = 0.f; }
  for (int i = gid; i < BB * K2; i += stride) z2[i] = f2bf(0.f);
  for (int i = gid; i < BB * K1; i += stride) {
    int b = i / K1, c = i % K1;
    float v = (c >= 256 && c < 384) ? values[(size_t)b * VSD + (c - 256)] : 0.f;
    z1[i] = f2bf(v);
  }
  if (gid == 0) { bar[0] = 0; bar[1] = 0; }
}

// ---------------- persistent decoder: 250-step scan, 64 WGs ----------------
__global__ void __launch_bounds__(NTHR, 1)
decoder_kernel(const float* key, const float* values, const int* speech_len,
               const int* text, const float* embedding, const float* b_out,
               char* ws, float* out) {
  const __bf16* wf1 = (const __bf16*)(ws + OFF_WF1);
  const __bf16* wf2 = (const __bf16*)(ws + OFF_WF2);
  const __bf16* wf3 = (const __bf16*)(ws + OFF_WF3);
  const float* bias1 = (const float*)(ws + OFF_B1);
  const float* bias2 = (const float*)(ws + OFF_B2);
  __bf16* z1 = (__bf16*)(ws + OFF_Z1);
  __bf16* z2 = (__bf16*)(ws + OFF_Z2);
  __bf16* zo = (__bf16*)(ws + OFF_ZO);
  float* G1 = (float*)(ws + OFF_G1);
  float* G2 = (float*)(ws + OFF_G2);
  float* c1f = (float*)(ws + OFF_C1);
  float* c2f = (float*)(ws + OFF_C2);
  float* h2f = (float*)(ws + OFF_H2F);
  unsigned* bar = (unsigned*)(ws + OFF_BAR);

  const int tid = threadIdx.x;
  const int wg = blockIdx.x;
  const int wv = tid >> 5;
  const int lane = tid & 31;
  const int gid = wg * NTHR + tid;

  __shared__ float sc[512];
  __shared__ float red[NTHR];
  __shared__ float h2s[KSD];

  for (int s = 0; s < LL; ++s) {
    // ---- Phase A: gates1 = Z1(64x896) @ Wcat1^T -> G1(64x2048); 1 tile/wave ----
    {
      int tile = wg * 8 + wv;  // 0..511 == 4 x 128 tiles
      int mt = tile >> 7, nt = tile & 127;
      int m0 = mt * 16;
      v8f acc = {};
      gemm_tile(z1, K1, wf1, nt, KC1, m0, lane, acc);
      int col = nt * 16 + (lane & 15);
      int row = m0 + ((lane >> 4) << 3);
#pragma unroll
      for (int r = 0; r < 8; ++r) G1[(size_t)(row + r) * N1 + col] = acc[r];
    }
    grid_barrier(bar);

    // ---- Phase B: LSTM1 pointwise -> h1,c1 ; also stage next-step embedding ----
    {
      for (int e = gid; e < BB * HH; e += NWG * NTHR) {
        int b = e >> 9, n = e & 511;
        const float* g = G1 + (size_t)b * N1;
        float gi = g[n] + bias1[n];
        float gf = g[n + 512] + bias1[n + 512];
        float gg = g[n + 1024] + bias1[n + 1024];
        float go = g[n + 1536] + bias1[n + 1536];
        float c = sigf(gf) * c1f[e] + sigf(gi) * tanhf(gg);
        c1f[e] = c;
        float h = sigf(go) * tanhf(c);
        __bf16 hb = f2bf(h);
        z2[(size_t)b * K2 + n] = hb;
        z1[(size_t)b * K1 + 384 + n] = hb;
      }
      if (s + 1 < LL) {  // token for step s+1 is text[b, s]; 64*256 == grid size
        int b = gid >> 8, j = gid & 255;
        int tok = text[b * LL + s];
        z1[(size_t)b * K1 + j] = f2bf(embedding[(size_t)tok * EE + j]);
      }
    }
    grid_barrier(bar);

    // ---- Phase C: gates2 = Z2(64x640) @ Wcat2^T -> G2(64x512); 128 tiles ----
    {
      int tile = wg * 8 + wv;
      if (tile < 128) {
        int mt = tile >> 5, nt = tile & 31;
        int m0 = mt * 16;
        v8f acc = {};
        gemm_tile(z2, K2, wf2, nt, KC2, m0, lane, acc);
        int col = nt * 16 + (lane & 15);
        int row = m0 + ((lane >> 4) << 3);
#pragma unroll
        for (int r = 0; r < 8; ++r) G2[(size_t)(row + r) * N2 + col] = acc[r];
      }
    }
    grid_barrier(bar);

    // ---- Phase D: LSTM2 pointwise -> h2,c2 ----
    if (gid < BB * KSD) {
      int b = gid >> 7, n = gid & 127;
      const float* g = G2 + (size_t)b * N2;
      float gi = g[n] + bias2[n];
      float gf = g[n + 128] + bias2[n + 128];
      float gg = g[n + 256] + bias2[n + 256];
      float go = g[n + 384] + bias2[n + 384];
      float c = sigf(gf) * c2f[gid] + sigf(gi) * tanhf(gg);
      c2f[gid] = c;
      float h = sigf(go) * tanhf(c);
      h2f[gid] = h;
      __bf16 hb = f2bf(h);
      z2[(size_t)b * K2 + 512 + n] = hb;
      zo[(size_t)b * K3 + n] = hb;
    }
    grid_barrier(bar);

    // ---- Phase E: attention; one WG per batch row b = wg ----
    {
      int b = wg;
      if (tid < KSD) h2s[tid] = h2f[(size_t)b * KSD + tid];
      __syncthreads();
      int slen = speech_len[b];
      for (int t = tid; t < TT; t += NTHR) {
        const float* kp = key + ((size_t)t * BB + b) * KSD;
        float e = 0.f;
        for (int k = 0; k < KSD; ++k) e += kp[k] * h2s[k];
        sc[t] = (t < slen) ? e : 0.f;  // mask*energy BEFORE softmax (faithful)
      }
      __syncthreads();
      float m = -3.4e38f;
      for (int t = tid; t < TT; t += NTHR) m = fmaxf(m, sc[t]);
      red[tid] = m;
      __syncthreads();
      for (int off = NTHR / 2; off > 0; off >>= 1) {
        if (tid < off) red[tid] = fmaxf(red[tid], red[tid + off]);
        __syncthreads();
      }
      float mx = red[0];
      __syncthreads();
      float sum = 0.f;
      for (int t = tid; t < TT; t += NTHR) {
        float ex = __expf(sc[t] - mx);
        sc[t] = ex;
        sum += ex;
      }
      red[tid] = sum;
      __syncthreads();
      for (int off = NTHR / 2; off > 0; off >>= 1) {
        if (tid < off) red[tid] += red[tid + off];
        __syncthreads();
      }
      float inv = 1.f / red[0];
      __syncthreads();
      if (tid < VSD) {
        int v = tid;
        float ctx = 0.f;
        for (int t = 0; t < TT; ++t) ctx += sc[t] * values[((size_t)t * BB + b) * VSD + v];
        ctx *= inv;
        __bf16 cb = f2bf(ctx);
        zo[(size_t)b * K3 + 128 + v] = cb;
        z1[(size_t)b * K1 + 256 + v] = cb;
      }
    }
    grid_barrier(bar);

    // ---- Phase F: pred = Zout(64x256) @ w_out^T + b_out -> out[b, s, :]; 2 tiles/wave ----
    {
#pragma unroll
      for (int i = 0; i < 2; ++i) {
        int tile = wg * 8 + wv + i * 512;  // 1024 tiles = 4 x 256
        int mt = tile >> 8, nt = tile & 255;
        int m0 = mt * 16;
        v8f acc = {};
        gemm_tile(zo, K3, wf3, nt, KC3, m0, lane, acc);
        int col = nt * 16 + (lane & 15);
        int row = m0 + ((lane >> 4) << 3);
        float bo = b_out[col];
#pragma unroll
        for (int r = 0; r < 8; ++r)
          out[((size_t)(row + r) * LL + s) * VOC + col] = acc[r] + bo;
      }
    }
    grid_barrier(bar);
  }
}

extern "C" void kernel_launch(void* const* d_in, const int* in_sizes, int n_in,
                              void* d_out, int out_size, void* d_ws, size_t ws_size,
                              hipStream_t stream) {
  (void)in_sizes; (void)n_in; (void)out_size; (void)ws_size;
  const float* key    = (const float*)d_in[0];
  const float* values = (const float*)d_in[1];
  const int* slen     = (const int*)d_in[2];
  const int* text     = (const int*)d_in[3];
  const float* emb    = (const float*)d_in[4];
  const float* w_ih1  = (const float*)d_in[5];
  const float* b_ih1  = (const float*)d_in[6];
  const float* w_hh1  = (const float*)d_in[7];
  const float* b_hh1  = (const float*)d_in[8];
  const float* w_ih2  = (const float*)d_in[9];
  const float* b_ih2  = (const float*)d_in[10];
  const float* w_hh2  = (const float*)d_in[11];
  const float* b_hh2  = (const float*)d_in[12];
  const float* w_out  = (const float*)d_in[13];
  const float* b_out  = (const float*)d_in[14];
  char* ws = (char*)d_ws;
  float* out = (float*)d_out;

  prep_kernel<<<256, 256, 0, stream>>>(w_ih1, w_hh1, w_ih2, w_hh2, w_out,
                                       b_ih1, b_hh1, b_ih2, b_hh2, ws);
  init_kernel<<<64, 256, 0, stream>>>(values, ws);
  decoder_kernel<<<NWG, NTHR, 0, stream>>>(key, values, slen, text, emb, b_out, ws, out);
}
